// Router_80642305950274
// MI455X (gfx1250) — matrix operand order repaired and checked by reading, not compiled, not run on hardware
//
#include <hip/hip_runtime.h>
#include <math.h>

// ---------------------------------------------------------------------------
// MoE router top-2 for MI455X (gfx1250).
//
// Roofline: T=32768 tokens, H=2048, E=64 -> 8.6 GFLOP vs 268 MB HBM reads
// (x stream dominates). Memory floor ~11.5 us @ 23.3 TB/s => use bf16 WMMA
// (16x16x32) with an error-compensated hi/lo split (xh*wh + xh*wl + xl*wh,
// ~fp32 accuracy) — compute stays far under the memory roofline.
//
// Data path:
//   prep kernel : W (512 KB fp32, L2-resident) -> bf16 hi/lo planes in d_ws,
//                 done ONCE per launch instead of per-workgroup.
//   main kernel : x tile staged via LDS (fp32->bf16 hi/lo, padded rows),
//                 B fragments loaded as contiguous 32 B runs straight from
//                 L2 into registers (lane k order 16*half+e, as in A, = row-major W bf16).
// ---------------------------------------------------------------------------

typedef unsigned short u16t;
typedef __attribute__((ext_vector_type(8)))  u16t   v8u;
typedef __attribute__((ext_vector_type(16))) u16t   v16u;
typedef __attribute__((ext_vector_type(16))) __bf16 v16bf;
typedef __attribute__((ext_vector_type(8)))  float  v8f;

union V16Pun { v16u u; v16bf b; };

#define KC   64     // K-chunk per iteration (2 WMMA k-steps)
#define PAD  72     // LDS row pitch in bf16 elems (144 B: breaks bank conflicts)
#define NEXP 64     // experts (4 waves x 16)

__device__ __forceinline__ u16t bf16_rne(unsigned u) {
  // IEEE round-to-nearest-even fp32 -> bf16 (matches WMMA RNE semantics)
  return (u16t)((u + 0x7FFFu + ((u >> 16) & 1u)) >> 16);
}

__device__ __forceinline__ void split_hi_lo(float f, u16t &hi, u16t &lo) {
  unsigned u = __float_as_uint(f);
  u16t h     = bf16_rne(u);
  float fh   = __uint_as_float(((unsigned)h) << 16);
  float fl   = f - fh;                  // exact residual
  hi = h;
  lo = bf16_rne(__float_as_uint(fl));
}

// A fragment: 16x32 bf16, lane = (half<<4)|m.
// Element e (0..15): K = ks + 16*half + e  -> two 16 B runs.
__device__ __forceinline__ v16bf fragA(const u16t* __restrict__ row, int ks, int half) {
  v8u p0 = *(const v8u*)(row + ks + 16 * half);
  v8u p1 = *(const v8u*)(row + ks + 16 * half + 8);
  V16Pun r;
#pragma unroll
  for (int i = 0; i < 8; ++i) { r.u[i] = p0[i]; r.u[i + 8] = p1[i]; }
  return r.b;
}

// B fragment: 32x16 bf16, lane = (khalf<<4)|n.
// Element e (0..15): K = ks + 16*khalf + e -> one contiguous 32 B run.
// Works on LDS or global pointers (address space inferred after inlining).
__device__ __forceinline__ v16bf fragB(const u16t* __restrict__ p) {
  v8u p0 = *(const v8u*)(p);
  v8u p1 = *(const v8u*)(p + 8);
  V16Pun r;
#pragma unroll
  for (int i = 0; i < 8; ++i) { r.u[i] = p0[i]; r.u[i + 8] = p1[i]; }
  return r.b;
}

// ---------------------------------------------------------------------------
// Shared epilogue: bias add, logits -> LDS, per-token top-2 + closed-form
// renormalized softmax:  softmax->top2->renorm == w1 = 1/(1+exp(l2-l1)).
// Strict '>' reproduces jax.lax.top_k lowest-index tie-breaking.
// ---------------------------------------------------------------------------
__device__ __forceinline__ void router_epilogue(v8f acc, float lg[16][65],
                                                const float* __restrict__ bvec,
                                                float* __restrict__ out,
                                                int T, long tok0,
                                                int t, int half, int ncol) {
  const float bias = bvec[ncol];
#pragma unroll
  for (int r = 0; r < 8; ++r)            // C layout: VGPR r -> M = r + 8*half
    lg[r + 8 * half][ncol] = acc[r] + bias;
  __syncthreads();

  if (t < 16) {
    const float* row = &lg[t][0];
    float b1 = -__builtin_inff(); int i1 = 0;
#pragma unroll 4
    for (int e = 0; e < NEXP; ++e) {
      float v = row[e];
      if (v > b1) { b1 = v; i1 = e; }
    }
    float b2 = -__builtin_inff(); int i2 = 0;
#pragma unroll 4
    for (int e = 0; e < NEXP; ++e) {
      if (e == i1) continue;
      float v = row[e];
      if (v > b2) { b2 = v; i2 = e; }
    }
    float ex = expf(b2 - b1);
    float w1 = 1.0f / (1.0f + ex);
    float w2 = ex * w1;
    long tok = tok0 + t;
    out[tok * 2 + 0] = w1;
    out[tok * 2 + 1] = w2;
    float* oid = out + 2 * (long)T;      // second tuple output (expert ids)
    oid[tok * 2 + 0] = (float)i1;
    oid[tok * 2 + 1] = (float)i2;
  }
}

// ---------------------------------------------------------------------------
// Prep: split W fp32 -> bf16 hi/lo planes once (vector float4 stream).
// ---------------------------------------------------------------------------
__global__ __launch_bounds__(256)
void Router_prep_w_kernel(const float* __restrict__ W,
                          u16t* __restrict__ Wh, u16t* __restrict__ Wl,
                          long n4) {
  long i = (long)blockIdx.x * blockDim.x + threadIdx.x;
  if (i >= n4) return;
  const float4 v = *(const float4*)(W + i * 4);
  u16t h0, l0, h1, l1, h2, l2, h3, l3;
  split_hi_lo(v.x, h0, l0); split_hi_lo(v.y, h1, l1);
  split_hi_lo(v.z, h2, l2); split_hi_lo(v.w, h3, l3);
  v8u* dh = (v8u*)(Wh + i * 4);   // write 4x u16 (8 B); use ushort4-style store
  ushort4 ph = make_ushort4(h0, h1, h2, h3);
  ushort4 pl = make_ushort4(l0, l1, l2, l3);
  *(ushort4*)(Wh + i * 4) = ph;
  *(ushort4*)(Wl + i * 4) = pl;
  (void)dh;
}

// ---------------------------------------------------------------------------
// Fast path: pre-split W in d_ws. B fragments straight from L2, A via LDS.
// ---------------------------------------------------------------------------
__global__ __launch_bounds__(128)
void Router_80642305950274_main(const float* __restrict__ x,
                                const u16t* __restrict__ Wh,
                                const u16t* __restrict__ Wl,
                                const float* __restrict__ bvec,
                                float* __restrict__ out,
                                int T, int H) {
  __shared__ alignas(16) u16t Ah[16][PAD];
  __shared__ alignas(16) u16t Al[16][PAD];
  __shared__ float lg[16][65];

  const int  t    = threadIdx.x;        // 0..127 (4 waves, wave32)
  const int  lane = t & 31;
  const int  wv   = t >> 5;             // wave -> 16-expert N-tile
  const int  half = lane >> 4;
  const int  mrow = lane & 15;
  const int  ncol = (wv << 4) + (lane & 15);   // global expert 0..63
  const long tok0 = (long)blockIdx.x * 16;

  const u16t* __restrict__ whRow = Wh + (long)ncol * H + 16 * half;
  const u16t* __restrict__ wlRow = Wl + (long)ncol * H + 16 * half;

  v8f acc = {};

  for (int k0 = 0; k0 < H; k0 += KC) {
    // ---- stage x tile: 16 tokens x 64 k fp32 -> bf16 hi/lo in LDS ----------
#pragma unroll
    for (int i = 0; i < 2; ++i) {
      int s   = t + i * 128;            // 256 float4 slots
      int row = s >> 4;                 // token row 0..15
      int c4  = (s & 15) << 2;          // k offset 0..60
      const float4 v = *(const float4*)(x + (tok0 + row) * (long)H + k0 + c4);
      u16t h, l;
      split_hi_lo(v.x, h, l); Ah[row][c4 + 0] = h; Al[row][c4 + 0] = l;
      split_hi_lo(v.y, h, l); Ah[row][c4 + 1] = h; Al[row][c4 + 1] = l;
      split_hi_lo(v.z, h, l); Ah[row][c4 + 2] = h; Al[row][c4 + 2] = l;
      split_hi_lo(v.w, h, l); Ah[row][c4 + 3] = h; Al[row][c4 + 3] = l;
    }
    // prefetch next x chunk toward WGP$ while computing on this one
    if (k0 + KC < H)
      __builtin_prefetch(x + (tok0 + (t >> 4)) * (long)H + (k0 + KC) + ((t & 15) << 2), 0, 1);

    __syncthreads();

    // ---- 2 k-steps x 3 compensated WMMAs: D += Ah*Bh + Ah*Bl + Al*Bh -------
    // B fragments are contiguous 32 B runs in L2-resident bf16 W planes.
#pragma unroll
    for (int ks = 0; ks < KC; ks += 32) {
      v16bf a_h = fragA(&Ah[mrow][0], ks, half);
      v16bf a_l = fragA(&Al[mrow][0], ks, half);
      v16bf b_h = fragB(whRow + k0 + ks);
      v16bf b_l = fragB(wlRow + k0 + ks);
      acc = __builtin_amdgcn_wmma_f32_16x16x32_bf16(false, a_h, false, b_h,
                                                    (short)0, acc, false, false);
      acc = __builtin_amdgcn_wmma_f32_16x16x32_bf16(false, a_h, false, b_l,
                                                    (short)0, acc, false, false);
      acc = __builtin_amdgcn_wmma_f32_16x16x32_bf16(false, a_l, false, b_h,
                                                    (short)0, acc, false, false);
    }
    __syncthreads();
  }

  router_epilogue(acc, lg, bvec, out, T, tok0, t, half, ncol);
}

// ---------------------------------------------------------------------------
// Fallback (ws too small): self-converting variant, W staged via LDS.
// ---------------------------------------------------------------------------
__global__ __launch_bounds__(128)
void Router_80642305950274_fallback(const float* __restrict__ x,
                                    const float* __restrict__ Wm,
                                    const float* __restrict__ bvec,
                                    float* __restrict__ out,
                                    int T, int H) {
  __shared__ alignas(16) u16t Ah[16][PAD];
  __shared__ alignas(16) u16t Al[16][PAD];
  __shared__ alignas(16) u16t Bh[NEXP][PAD];
  __shared__ alignas(16) u16t Bl[NEXP][PAD];
  __shared__ float lg[16][65];

  const int  t    = threadIdx.x;
  const int  lane = t & 31;
  const int  wv   = t >> 5;
  const int  half = lane >> 4;
  const int  mrow = lane & 15;
  const int  ncol = (wv << 4) + (lane & 15);
  const long tok0 = (long)blockIdx.x * 16;

  v8f acc = {};

  for (int k0 = 0; k0 < H; k0 += KC) {
#pragma unroll
    for (int i = 0; i < 2; ++i) {
      int s = t + i * 128, row = s >> 4, c4 = (s & 15) << 2;
      const float4 v = *(const float4*)(x + (tok0 + row) * (long)H + k0 + c4);
      u16t h, l;
      split_hi_lo(v.x, h, l); Ah[row][c4 + 0] = h; Al[row][c4 + 0] = l;
      split_hi_lo(v.y, h, l); Ah[row][c4 + 1] = h; Al[row][c4 + 1] = l;
      split_hi_lo(v.z, h, l); Ah[row][c4 + 2] = h; Al[row][c4 + 2] = l;
      split_hi_lo(v.w, h, l); Ah[row][c4 + 3] = h; Al[row][c4 + 3] = l;
    }
#pragma unroll
    for (int i = 0; i < 8; ++i) {
      int s = t + i * 128, row = s >> 4, c4 = (s & 15) << 2;
      const float4 v = *(const float4*)(Wm + (long)row * H + k0 + c4);
      u16t h, l;
      split_hi_lo(v.x, h, l); Bh[row][c4 + 0] = h; Bl[row][c4 + 0] = l;
      split_hi_lo(v.y, h, l); Bh[row][c4 + 1] = h; Bl[row][c4 + 1] = l;
      split_hi_lo(v.z, h, l); Bh[row][c4 + 2] = h; Bl[row][c4 + 2] = l;
      split_hi_lo(v.w, h, l); Bh[row][c4 + 3] = h; Bl[row][c4 + 3] = l;
    }
    __syncthreads();
#pragma unroll
    for (int ks = 0; ks < KC; ks += 32) {
      v16bf a_h = fragA(&Ah[mrow][0], ks, half);
      v16bf a_l = fragA(&Al[mrow][0], ks, half);
      v16bf b_h = fragB(&Bh[ncol][ks + 16 * half]);
      v16bf b_l = fragB(&Bl[ncol][ks + 16 * half]);
      acc = __builtin_amdgcn_wmma_f32_16x16x32_bf16(false, a_h, false, b_h,
                                                    (short)0, acc, false, false);
      acc = __builtin_amdgcn_wmma_f32_16x16x32_bf16(false, a_h, false, b_l,
                                                    (short)0, acc, false, false);
      acc = __builtin_amdgcn_wmma_f32_16x16x32_bf16(false, a_l, false, b_h,
                                                    (short)0, acc, false, false);
    }
    __syncthreads();
  }

  router_epilogue(acc, lg, bvec, out, T, tok0, t, half, ncol);
}

extern "C" void kernel_launch(void* const* d_in, const int* in_sizes, int n_in,
                              void* d_out, int out_size, void* d_ws, size_t ws_size,
                              hipStream_t stream) {
  const float* x  = (const float*)d_in[0];   // [B,S,H] fp32
  const float* Wm = (const float*)d_in[1];   // [E,H]   fp32
  const float* bv = (const float*)d_in[2];   // [E]     fp32
  float* out = (float*)d_out;                // [T*2 weights | T*2 expert ids]

  const int  E  = in_sizes[2];               // 64
  const int  H  = in_sizes[1] / E;           // 2048
  const int  T  = in_sizes[0] / H;           // 32768
  const long EH = (long)E * H;

  dim3 grid(T / 16);                         // 16 tokens per workgroup
  dim3 block(128);                           // 4 wave32s (one 16-expert tile each)

  const size_t need = (size_t)EH * 2 * sizeof(u16t);  // hi + lo planes
  if (ws_size >= need) {
    u16t* Wh = (u16t*)d_ws;
    u16t* Wl = Wh + EH;
    long n4 = EH / 4;
    Router_prep_w_kernel<<<dim3((unsigned)((n4 + 255) / 256)), dim3(256), 0, stream>>>(
        Wm, Wh, Wl, n4);
    Router_80642305950274_main<<<grid, block, 0, stream>>>(x, Wh, Wl, bv, out, T, H);
  } else {
    Router_80642305950274_fallback<<<grid, block, 0, stream>>>(x, Wm, bv, out, T, H);
  }
}
